// MultiHeadSelfAttention_49168785604666
// MI455X (gfx1250) — matrix operand order repaired and checked by reading, not compiled, not run on hardware
//
#include <hip/hip_runtime.h>
#include <hip/hip_bf16.h>

#define H_NUM  16
#define D_HEAD 64
#define C_HID  1024
#define F3     3072

typedef __bf16 bf16;
typedef __attribute__((ext_vector_type(16))) __bf16 v16bf;
typedef __attribute__((ext_vector_type(8)))  __bf16 v8bf;
typedef __attribute__((ext_vector_type(8)))  float  v8f;
typedef __attribute__((ext_vector_type(4)))  unsigned int u32x4;
typedef __attribute__((ext_vector_type(8)))  int i32x8;
typedef __attribute__((ext_vector_type(4)))  int i32x4;

#if defined(__has_builtin)
#  if __has_builtin(__builtin_amdgcn_tensor_load_to_lds)
#    define HAVE_TDM 1
#  endif
#  if __has_builtin(__builtin_amdgcn_s_wait_tensorcnt)
#    define HAVE_TENSORCNT 1
#  endif
#endif

__device__ __forceinline__ bf16 f2bf(float f) {
  unsigned u = __builtin_bit_cast(unsigned, f);
  unsigned r = u + 0x7FFFu + ((u >> 16) & 1u);   // round-to-nearest-even
  unsigned short h = (unsigned short)(r >> 16);
  return __builtin_bit_cast(bf16, h);
}
__device__ __forceinline__ float bf2f(bf16 h) {
  unsigned u = ((unsigned)__builtin_bit_cast(unsigned short, h)) << 16;
  return __builtin_bit_cast(float, u);
}

__device__ __forceinline__ v8f wmma_bf16(v16bf a, v16bf b, v8f c) {
  // (neg_a, A, neg_b, B, c_mod, C, reuse_a, reuse_b)
  return __builtin_amdgcn_wmma_f32_16x16x32_bf16(false, a, false, b, (short)0, c, false, false);
}

// Issue one TDM load of 2*`nElems` contiguous bytes (as 1-D tensor of 2-byte
// elements) from global `gaddr` to LDS offset `ldsOff`.  Wave-level op,
// tracked by TENSORcnt.
__device__ __forceinline__ void tdm_load_1d(unsigned long long gaddr, unsigned ldsOff,
                                            unsigned nElems) {
#if defined(HAVE_TDM)
  u32x4 g0;
  g0[0] = 1u;                                    // count=1, user descriptor
  g0[1] = ldsOff;                                // lds_addr (bytes)
  g0[2] = (unsigned)(gaddr & 0xFFFFFFFFull);     // global_addr[31:0]
  g0[3] = (unsigned)((gaddr >> 32) & 0x1FFFFFFull) | (2u << 30);  // addr[56:32] | type=2
  i32x8 g1;
  g1[0] = (int)(1u << 16);                       // workgroup_mask=0, data_size=1 (2B)
  g1[1] = (int)((nElems & 0xFFFFu) << 16);       // tensor_dim0[15:0]
  g1[2] = (int)((nElems >> 16) & 0xFFFFu);       // tensor_dim0[31:16], tensor_dim1=0
  g1[3] = (int)((nElems & 0xFFFFu) << 16);       // tile_dim0 = nElems (1-D tile)
  g1[4] = 0;                                     // tile_dim1=0, tile_dim2=0
  g1[5] = 0;                                     // tensor_dim0_stride (unused, 1-D)
  g1[6] = 0;
  g1[7] = 0;
  i32x4 g2 = {0, 0, 0, 0};
  i32x4 g3 = {0, 0, 0, 0};
#if __clang_major__ >= 23
  i32x8 g4 = {0, 0, 0, 0, 0, 0, 0, 0};
  __builtin_amdgcn_tensor_load_to_lds(g0, g1, g2, g3, g4, 0);
#else
  __builtin_amdgcn_tensor_load_to_lds(g0, g1, g2, g3, 0);
#endif
#endif
}

__device__ __forceinline__ void wait_tensorcnt0() {
#if defined(HAVE_TENSORCNT)
  __builtin_amdgcn_s_wait_tensorcnt(0);
#else
  asm volatile("s_wait_tensorcnt 0x0" ::: "memory");
#endif
}

// ---------------------------------------------------------------- convert
__global__ void cvt_f32_bf16(const float* __restrict__ in, bf16* __restrict__ out, int n) {
  int i = blockIdx.x * blockDim.x + threadIdx.x;
  if (i < n) out[i] = f2bf(in[i]);
}

// ---------------------------------------------------------------- GEMM
// C[M,N] = A[M,K] * W[N,K]^T   (both bf16, K-contiguous; f32 accumulate)
// A-tile (16 rows x full K) staged in LDS once per block via TDM, shared by
// the 4 waves; each wave computes a 16x64 output tile.
__device__ __forceinline__ void storeOut(float* p, float v) { *p = v; }
__device__ __forceinline__ void storeOut(bf16*  p, float v) { *p = f2bf(v); }

template <typename OutT>
__global__ __launch_bounds__(128) void gemm_bf16_wmma(
    const bf16* __restrict__ A, const bf16* __restrict__ W,
    OutT* __restrict__ Cout, int M, int N, int K) {
  __shared__ alignas(128) bf16 aTile[16 * C_HID];   // 32 KB (K <= C_HID)
  const int lane = threadIdx.x & 31;
  const int wave = threadIdx.x >> 5;
  const int half = lane >> 4;
  const int l16  = lane & 15;
  const int mBase = blockIdx.y * 16;
  const int nBase = blockIdx.x * 256 + wave * 64;   // each wave: 16x64 tile

  // ---- stage A tile (rows mBase..mBase+15, full K: contiguous 16*K elems)
#if defined(HAVE_TDM)
  if (wave == 0) {
    unsigned ldsOff = (unsigned)(size_t)(&aTile[0]);   // generic->LDS offset (addr[31:0])
    unsigned long long ga = (unsigned long long)(size_t)(A + (size_t)mBase * K);
    tdm_load_1d(ga, ldsOff, (unsigned)(16 * K));
    wait_tensorcnt0();
  }
#else
  {
    const bf16* src = A + (size_t)mBase * K;
    for (int idx = threadIdx.x * 8; idx < 16 * K; idx += 128 * 8)
      *(v8bf*)(aTile + idx) = *(const v8bf*)(src + idx);
  }
#endif
  __syncthreads();

  v8f acc[4] = {{}, {}, {}, {}};

  const bf16* ArowLds = aTile + (size_t)l16 * K;
  for (int k = 0; k < K; k += 32) {
    // A-frag (16x32 bf16): lane<16 K=0..15, lane>=16 K=16..31
    v8bf alo = *(const v8bf*)(ArowLds + k + 16 * half);       // ds_load_b128
    v8bf ahi = *(const v8bf*)(ArowLds + k + 8 + 16 * half);  // ds_load_b128
    v16bf a;
#pragma unroll
    for (int i = 0; i < 8; ++i) { a[i] = alo[i]; a[i + 8] = ahi[i]; }
#pragma unroll
    for (int j = 0; j < 4; ++j) {
      // B-frag (32x16): lane holds column n=l16 (+16 cols per j), K = 16*half + 0..15
      const bf16* Wrow = W + (size_t)(nBase + j * 16 + l16) * K;
      v16bf b = *(const v16bf*)(Wrow + k + 16 * half);
      __builtin_prefetch((const void*)(Wrow + k + 64 + 16 * half), 0, 3);
      acc[j] = wmma_bf16(a, b, acc[j]);
    }
  }
#pragma unroll
  for (int j = 0; j < 4; ++j) {
#pragma unroll
    for (int i = 0; i < 8; ++i) {
      int row = mBase + i + 8 * half;     // C layout: VGPR i, half selects row block
      int col = nBase + j * 16 + l16;
      storeOut(&Cout[(size_t)row * N + col], acc[j][i]);
    }
  }
}

// ---------------------------------------------------------------- RoPE + head split
// qkv[B,T,3C] -> qh[B,H,T,D], kh[B,H,T,D], vt[B,H,D,T] (V transposed for P*V B-operand)
__global__ void rope_split_kernel(const bf16* __restrict__ qkv,
                                  const float* __restrict__ cosT, const float* __restrict__ sinT,
                                  bf16* __restrict__ qh, bf16* __restrict__ kh, bf16* __restrict__ vt,
                                  int B, int T) {
  int d = threadIdx.x;          // 0..63
  int t = blockIdx.x;
  int h = blockIdx.y;
  int b = blockIdx.z;
  size_t base = ((size_t)(b * T + t)) * F3 + (size_t)h * D_HEAD;
  float c = cosT[t * D_HEAD + d];
  float s = sinT[t * D_HEAD + d];
  float q = bf2f(qkv[base + d]);
  float k = bf2f(qkv[base + C_HID + d]);
  float v = bf2f(qkv[base + 2 * C_HID + d]);
  int   dp  = (d < 32) ? d + 32 : d - 32;
  float sgn = (d < 32) ? -1.0f : 1.0f;
  float qp = bf2f(qkv[base + dp]);
  float kp = bf2f(qkv[base + C_HID + dp]);
  float qo = q * c + sgn * qp * s;
  float ko = k * c + sgn * kp * s;
  size_t bh = (size_t)(b * H_NUM + h);
  qh[(bh * T + t) * D_HEAD + d] = f2bf(qo);
  kh[(bh * T + t) * D_HEAD + d] = f2bf(ko);
  vt[(bh * D_HEAD + d) * T + t] = f2bf(v);
}

// ---------------------------------------------------------------- flash attention
// One wave handles a 16-query tile of one (b,h); 32 keys per step.
__global__ __launch_bounds__(128) void attn_fa_kernel(
    const bf16* __restrict__ qh, const bf16* __restrict__ kh, const bf16* __restrict__ vt,
    bf16* __restrict__ obf, int B, int T) {
  __shared__ alignas(32) bf16 pbuf[4][16 * 32];   // per-wave probs staging (C-layout -> A-layout)
  const int lane = threadIdx.x & 31;
  const int wave = threadIdx.x >> 5;
  const int half = lane >> 4;
  const int l16  = lane & 15;
  const int h = blockIdx.y, b = blockIdx.z;
  const int qBase = (blockIdx.x * 4 + wave) * 16;
  const size_t bh = (size_t)(b * H_NUM + h);
  const bf16* Q  = qh + bh * T * D_HEAD;
  const bf16* Kp = kh + bh * T * D_HEAD;
  const bf16* Vp = vt + bh * (size_t)D_HEAD * T;

  // Q A-frags for d 0..31 and 32..63 (kept resident)
  const bf16* Qrow = Q + (size_t)(qBase + l16) * D_HEAD;
  v8bf q00 = *(const v8bf*)(Qrow + 16 * half);
  v8bf q01 = *(const v8bf*)(Qrow + 8 + 16 * half);
  v8bf q10 = *(const v8bf*)(Qrow + 32 + 16 * half);
  v8bf q11 = *(const v8bf*)(Qrow + 40 + 16 * half);
  v16bf qa0, qa1;
#pragma unroll
  for (int i = 0; i < 8; ++i) { qa0[i] = q00[i]; qa0[i + 8] = q01[i];
                                qa1[i] = q10[i]; qa1[i + 8] = q11[i]; }

  v8f o[4] = {{}, {}, {}, {}};
  float mrow[8], lrow[8];
#pragma unroll
  for (int i = 0; i < 8; ++i) { mrow[i] = -3.0e38f; lrow[i] = 0.0f; }

  const float scale = 0.125f;   // 1/sqrt(64)
  bf16* pw = &pbuf[wave][0];

  for (int kt = 0; kt < qBase + 16; kt += 32) {
    // ---- scores: q[16x64] . k^T[64x32] -> two 16x16 tiles
    v8f sc0 = {}, sc1 = {};
    {
      const bf16* Kr0 = Kp + (size_t)(kt + l16) * D_HEAD;
      v16bf b0 = *(const v16bf*)(Kr0 + 16 * half);
      v16bf b1 = *(const v16bf*)(Kr0 + 32 + 16 * half);
      sc0 = wmma_bf16(qa0, b0, sc0);
      sc0 = wmma_bf16(qa1, b1, sc0);
      const bf16* Kr1 = Kp + (size_t)(kt + 16 + l16) * D_HEAD;
      v16bf c0 = *(const v16bf*)(Kr1 + 16 * half);
      v16bf c1 = *(const v16bf*)(Kr1 + 32 + 16 * half);
      sc1 = wmma_bf16(qa0, c0, sc1);
      sc1 = wmma_bf16(qa1, c1, sc1);
    }
    // ---- online softmax (row reductions across each 16-lane half)
    const int key0 = kt + l16;
    const int key1 = kt + 16 + l16;
#pragma unroll
    for (int i = 0; i < 8; ++i) {
      int qRow = qBase + i + 8 * half;
      float s0 = (key0 <= qRow) ? sc0[i] * scale : -3.0e38f;
      float s1 = (key1 <= qRow) ? sc1[i] * scale : -3.0e38f;
      float mx = fmaxf(s0, s1);
#pragma unroll
      for (int off = 1; off < 16; off <<= 1) mx = fmaxf(mx, __shfl_xor(mx, off, 16));
      float mNew  = fmaxf(mrow[i], mx);
      float alpha = __expf(mrow[i] - mNew);
      float p0 = __expf(s0 - mNew);
      float p1 = __expf(s1 - mNew);
      float ps = p0 + p1;
#pragma unroll
      for (int off = 1; off < 16; off <<= 1) ps += __shfl_xor(ps, off, 16);
      lrow[i] = lrow[i] * alpha + ps;
      mrow[i] = mNew;
      o[0][i] *= alpha; o[1][i] *= alpha; o[2][i] *= alpha; o[3][i] *= alpha;
      int m = i + 8 * half;                // C-layout row owned by this (vgpr, half)
      pw[m * 32 + l16]      = f2bf(p0);
      pw[m * 32 + 16 + l16] = f2bf(p1);
    }
    asm volatile("s_wait_dscnt 0" ::: "memory");   // probs visible in LDS
    // ---- reload probs as A-frag (16x32)
    v8bf plo = *(const v8bf*)(pw + l16 * 32 + 16 * half);
    v8bf phi = *(const v8bf*)(pw + l16 * 32 + 8 + 16 * half);
    v16bf pa;
#pragma unroll
    for (int i = 0; i < 8; ++i) { pa[i] = plo[i]; pa[i + 8] = phi[i]; }
    // ---- P[16x32] . V[32x64] -> four 16x16 accum tiles
#pragma unroll
    for (int j = 0; j < 4; ++j) {
      const bf16* Vr = Vp + (size_t)(j * 16 + l16) * T;   // row d of V^T, keys contiguous
      v16bf bv = *(const v16bf*)(Vr + kt + 16 * half);
      o[j] = wmma_bf16(pa, bv, o[j]);
    }
  }

  // ---- normalize + store to [B,T,C]
  float inv[8];
#pragma unroll
  for (int i = 0; i < 8; ++i) inv[i] = 1.0f / lrow[i];
#pragma unroll
  for (int j = 0; j < 4; ++j) {
#pragma unroll
    for (int i = 0; i < 8; ++i) {
      int t   = qBase + i + 8 * half;
      int col = h * D_HEAD + j * 16 + l16;
      obf[((size_t)(b * T + t)) * C_HID + col] = f2bf(o[j][i] * inv[i]);
    }
  }
}

// ---------------------------------------------------------------- launch
extern "C" void kernel_launch(void* const* d_in, const int* in_sizes, int n_in,
                              void* d_out, int out_size, void* d_ws, size_t ws_size,
                              hipStream_t stream) {
  const float* x     = (const float*)d_in[0];
  const float* cosT  = (const float*)d_in[1];
  const float* sinT  = (const float*)d_in[2];
  const float* w_qkv = (const float*)d_in[3];
  const float* w_out = (const float*)d_in[4];
  const int B = 2, T = 2048;

  // workspace slicing (256B aligned); total ~72 MB
  char* p = (char*)d_ws;
  auto take = [&](size_t elems) {
    bf16* r = (bf16*)p;
    p += ((elems * sizeof(bf16)) + 255) & ~(size_t)255;
    return r;
  };
  bf16* x_bf   = take((size_t)B * T * C_HID);
  bf16* wq_bf  = take((size_t)F3 * C_HID);
  bf16* wo_bf  = take((size_t)C_HID * C_HID);
  bf16* qkv_bf = take((size_t)B * T * F3);
  bf16* qh     = take((size_t)B * H_NUM * T * D_HEAD);
  bf16* kh     = take((size_t)B * H_NUM * T * D_HEAD);
  bf16* vt     = take((size_t)B * H_NUM * T * D_HEAD);
  bf16* obf    = take((size_t)B * T * C_HID);

  int n1 = B * T * C_HID;
  int n2 = F3 * C_HID;
  int n3 = C_HID * C_HID;
  cvt_f32_bf16<<<(n1 + 255) / 256, 256, 0, stream>>>(x, x_bf, n1);
  cvt_f32_bf16<<<(n2 + 255) / 256, 256, 0, stream>>>(w_qkv, wq_bf, n2);
  cvt_f32_bf16<<<(n3 + 255) / 256, 256, 0, stream>>>(w_out, wo_bf, n3);

  // qkv = x @ w_qkv^T : M=4096 N=3072 K=1024
  {
    dim3 grid(F3 / 256, (B * T) / 16);
    gemm_bf16_wmma<bf16><<<grid, 128, 0, stream>>>(x_bf, wq_bf, qkv_bf, B * T, F3, C_HID);
  }
  // RoPE + split into heads (V transposed)
  {
    dim3 grid(T, H_NUM, B);
    rope_split_kernel<<<grid, D_HEAD, 0, stream>>>(qkv_bf, cosT, sinT, qh, kh, vt, B, T);
  }
  // causal flash attention
  {
    dim3 grid(T / 64, H_NUM, B);   // 4 waves/block * 16 rows/wave = 64 rows/block
    attn_fa_kernel<<<grid, 128, 0, stream>>>(qh, kh, vt, obf, B, T);
  }
  // out = attn @ w_out^T : M=4096 N=1024 K=1024
  {
    dim3 grid(C_HID / 256, (B * T) / 16);
    gemm_bf16_wmma<float><<<grid, 128, 0, stream>>>(obf, wo_bf, (float*)d_out, B * T, C_HID, C_HID);
  }
}
